// DecoderBlock_58514634441304
// MI455X (gfx1250) — compile-verified, matrix operand order checked by reading, not run on hardware
//
#include <hip/hip_runtime.h>
#include <hip/hip_bf16.h>

// MI455X (gfx1250) decoder block: RMSNorm -> QKV(WMMA) -> RoPE -> FlashAttn(WMMA)
// -> Wo+res(WMMA) -> RMSNorm -> W1(WMMA) -> SwiGLU -> W2+res(WMMA).
// GEMMs: 2x2 WMMA register tiles per wave (32x32), B-tile staged to LDS via
// gfx1250 async global->LDS (ASYNCcnt), double buffered.

typedef __attribute__((ext_vector_type(16))) __bf16 v16bf;
typedef __attribute__((ext_vector_type(8)))  float  v8f;
typedef __attribute__((ext_vector_type(4)))  int    v4i;

#define T_B   2
#define T_S   2048
#define T_D   1024
#define T_H   16
#define T_DK  64
#define T_DFF 2048
#define T_M   (T_B * T_S)   // 4096 tokens

__device__ __forceinline__ v8f wmma_bf16(v16bf a, v16bf b, v8f c) {
  return __builtin_amdgcn_wmma_f32_16x16x32_bf16(
      /*neg_a=*/false, a, /*neg_b=*/false, b, /*c_mod=*/(short)0, c,
      /*reuse_a=*/false, /*reuse_b=*/false);
}

// ---- gfx1250 async global->LDS (guarded: falls back to plain LDS copy) -------
#if defined(__has_builtin)
#if __has_builtin(__builtin_amdgcn_global_load_async_to_lds_b128)
#define HAVE_ASYNC_LDS 1
#endif
#endif
#ifndef HAVE_ASYNC_LDS
#define HAVE_ASYNC_LDS 0
#endif

__device__ __forceinline__ void async_copy16(const __bf16* g, __bf16* l) {
#if HAVE_ASYNC_LDS
  // param 0: int4 __device__* (global), param 1: LDS dest; generic pointers get
  // implicit addrspace conversion in HIP builtin calls.
  __builtin_amdgcn_global_load_async_to_lds_b128((v4i*)g, (v4i*)l, 0, 0);
#else
  *(uint4*)l = *(const uint4*)g;
#endif
}

__device__ __forceinline__ void wait_async() {
#if HAVE_ASYNC_LDS
#if __has_builtin(__builtin_amdgcn_s_wait_asynccnt)
  __builtin_amdgcn_s_wait_asynccnt(0);
#else
  asm volatile("s_wait_asynccnt 0" ::: "memory");
#endif
#endif
}

// ---------------- weight transpose + f32->bf16 cast: out[n*K+k] = in[k*N+n] ----
__global__ void k_transpose_cvt(const float* __restrict__ in,
                                __bf16* __restrict__ out, int K, int N) {
  size_t idx = (size_t)blockIdx.x * blockDim.x + threadIdx.x;
  size_t total = (size_t)K * N;
  if (idx >= total) return;
  size_t k = idx / N, n = idx % N;
  out[n * (size_t)K + k] = (__bf16)in[idx];
}

// ---------------- RMSNorm: f32 row -> bf16 row --------------------------------
__global__ void k_rmsnorm(const float* __restrict__ x, const float* __restrict__ w,
                          __bf16* __restrict__ out) {
  int row = blockIdx.x;
  const float* xr = x + (size_t)row * T_D;
  float ss = 0.f;
  for (int i = threadIdx.x; i < T_D; i += 256) { float v = xr[i]; ss += v * v; }
  for (int off = 16; off; off >>= 1) ss += __shfl_xor(ss, off, 32);
  __shared__ float red[8];
  __shared__ float rinv_s;
  if ((threadIdx.x & 31) == 0) red[threadIdx.x >> 5] = ss;
  __syncthreads();
  if (threadIdx.x == 0) {
    float t = 0.f;
    for (int i = 0; i < 8; i++) t += red[i];
    rinv_s = rsqrtf(t * (1.0f / T_D) + 1e-6f);
  }
  __syncthreads();
  float rinv = rinv_s;
  __bf16* orow = out + (size_t)row * T_D;
  for (int i = threadIdx.x; i < T_D; i += 256)
    orow[i] = (__bf16)(xr[i] * rinv * w[i]);
}

// ---------------- WMMA GEMM: C[M,N] = A[M,K](bf16) x BT[N,K](bf16) ------------
// 8 waves/block as 4(M)x2(N); each wave a 32x32 tile (2x2 WMMA frags).
// Block macro-tile 128(M) x 64(N). BT tile (64 x 32 K-slice) is staged into
// LDS with async global->LDS b128, double buffered; A-frags load direct.
// mode 0: bf16 out[m*N+col] (+bias)           (FFN up-proj u)
// mode 1: bf16 -> [B,H,S,DK] (+bias)          (q, k)
// mode 2: bf16 -> [B,H,DK,S] (+bias)          (v transposed for flash PV)
// mode 3: f32 out = acc + bias + res[m*N+col] (Wo proj, FFN down-proj)
__global__ void k_gemm_wmma(const __bf16* __restrict__ A, const __bf16* __restrict__ BT,
                            const float* __restrict__ bias, const float* __restrict__ res,
                            float* __restrict__ outf, __bf16* __restrict__ outb,
                            int M, int N, int K, int mode) {
  int lane = threadIdx.x & 31, wave = threadIdx.x >> 5;
  int wr = wave & 3;                 // wave row -> 32-row M subtile
  int wc = wave >> 2;                // wave col -> 32-col N subtile
  int nBlock = blockIdx.x * 64;
  int m0 = blockIdx.y * 128 + wr * 32;
  int n0 = nBlock + wc * 32;
  int half = lane >> 4, r = lane & 15;

  __shared__ __align__(16) __bf16 lb[2][64 * 32];   // 2 x 4KB B-tile buffers

  // each of 256 threads async-copies one 16B chunk of the 64x32 B tile
  int srow = threadIdx.x >> 2;                      // 0..63 (row of block N-tile)
  int schk = (threadIdx.x & 3) * 8;                 // bf16 offset of 16B chunk
  const __bf16* bsrc = BT + (size_t)(nBlock + srow) * K + schk;
  __bf16* bdst0 = &lb[0][srow * 32 + schk];
  __bf16* bdst1 = &lb[1][srow * 32 + schk];

  const __bf16* apr0 = A + (size_t)(m0 + r) * K + half * 16;       // rows 0..15
  const __bf16* apr1 = A + (size_t)(m0 + 16 + r) * K + half * 16;  // rows 16..31
  const __bf16* lbr0 = &lb[0][(wc * 32 + r) * 32 + half * 16];
  const __bf16* lbr1 = &lb[0][(wc * 32 + 16 + r) * 32 + half * 16];
  const int lbstride = 64 * 32;                     // buffer 1 offset (elements)

  v8f a00 = {}, a01 = {}, a10 = {}, a11 = {};

  async_copy16(bsrc, bdst0);                        // prime buffer 0
  int buf = 0;
  for (int k0 = 0; k0 < K; k0 += 32) {
    wait_async();
    __syncthreads();                                // B tile visible to all waves
    if (k0 + 32 < K)                                // prefetch next K-slice
      async_copy16(bsrc + k0 + 32, buf ? bdst0 : bdst1);
    __builtin_prefetch(apr0 + k0 + 256, 0, 1);
    __builtin_prefetch(apr1 + k0 + 256, 0, 1);
    v16bf fa0 = *(const v16bf*)(apr0 + k0);
    v16bf fa1 = *(const v16bf*)(apr1 + k0);
    v16bf fb0 = *(const v16bf*)(lbr0 + buf * lbstride);
    v16bf fb1 = *(const v16bf*)(lbr1 + buf * lbstride);
    a00 = wmma_bf16(fa0, fb0, a00);
    a01 = wmma_bf16(fa0, fb1, a01);
    a10 = wmma_bf16(fa1, fb0, a10);
    a11 = wmma_bf16(fa1, fb1, a11);
    __syncthreads();                                // reads done before overwrite
    buf ^= 1;
  }

  int col0 = n0 + r, col1 = n0 + 16 + r;
  float bv0 = bias ? bias[col0] : 0.f;
  float bv1 = bias ? bias[col1] : 0.f;
  v8f accs[2][2] = {{a00, a01}, {a10, a11}};
  float bvs[2] = {bv0, bv1};
  int cols[2] = {col0, col1};
#pragma unroll
  for (int si = 0; si < 2; si++) {
#pragma unroll
    for (int sj = 0; sj < 2; sj++) {
#pragma unroll
      for (int i = 0; i < 8; i++) {
        int m = m0 + si * 16 + i + 8 * half;  // lanes>=16 hold rows +8..+15
        int c = cols[sj];
        float v = accs[si][sj][i] + bvs[sj];
        if (mode == 0) {
          outb[(size_t)m * N + c] = (__bf16)v;
        } else if (mode == 1) {               // [B,H,S,DK], N==1024, S==2048
          int b = m >> 11, s = m & (T_S - 1);
          int hh = c >> 6, d = c & 63;
          outb[(((size_t)b * T_H + hh) * T_S + s) * T_DK + d] = (__bf16)v;
        } else if (mode == 2) {               // [B,H,DK,S]
          int b = m >> 11, s = m & (T_S - 1);
          int hh = c >> 6, d = c & 63;
          outb[(((size_t)b * T_H + hh) * T_DK + d) * T_S + s] = (__bf16)v;
        } else {
          outf[(size_t)m * N + c] = v + res[(size_t)m * N + c];
        }
      }
    }
  }
}

// ---------------- RoPE on q,k in [B,H,S,DK] bf16 ------------------------------
__global__ void k_rope(__bf16* __restrict__ q, __bf16* __restrict__ k) {
  size_t idx = (size_t)blockIdx.x * blockDim.x + threadIdx.x;  // B*H*S*32 threads
  int d = idx & 31;
  size_t t = idx >> 5;
  int s = t & (T_S - 1);
  size_t bh = t >> 11;
  float inv = __powf(10000.f, -(float)d * (1.0f / 32.0f));
  float ang = (float)s * inv;
  float sn, c;
  __sincosf(ang, &sn, &c);
  size_t base = (bh * T_S + s) * T_DK + d;
  float q1 = (float)q[base], q2 = (float)q[base + 32];
  q[base]      = (__bf16)(q1 * c - q2 * sn);
  q[base + 32] = (__bf16)(q2 * c + q1 * sn);
  float k1 = (float)k[base], k2 = (float)k[base + 32];
  k[base]      = (__bf16)(k1 * c - k2 * sn);
  k[base + 32] = (__bf16)(k2 * c + k1 * sn);
}

// ---------------- Flash attention (causal), one wave per 16-query tile --------
// q,k: [B,H,S,DK] bf16, vT: [B,H,DK,S] bf16, o: [B,S,D] bf16
__global__ void k_attn_flash(const __bf16* __restrict__ q, const __bf16* __restrict__ k,
                             const __bf16* __restrict__ vT, __bf16* __restrict__ o) {
  int lane = threadIdx.x & 31, wave = threadIdx.x >> 5;
  int bh = blockIdx.y;                       // b*H + h
  int qbase = (blockIdx.x * 4 + wave) * 16;
  int half = lane >> 4, r = lane & 15;

  const __bf16* qrow = q + ((size_t)bh * T_S + qbase + r) * T_DK;
  v16bf aq0 = *(const v16bf*)(qrow + half * 16);        // K = 0..31 of DK
  v16bf aq1 = *(const v16bf*)(qrow + 32 + half * 16);   // K = 32..63

  v8f oacc[4] = {};
  float mrun[8], lrun[8];
#pragma unroll
  for (int i = 0; i < 8; i++) { mrun[i] = -3.0e38f; lrun[i] = 0.f; }

  __shared__ __align__(64) __bf16 plds[4][16 * 32];     // per-wave P transpose buf
  __bf16* pl = plds[wave];

  int nkb = (qbase + 16 + 31) >> 5;                     // causal: keys <= qbase+15
  for (int kb = 0; kb < nkb; kb++) {
    int kbase = kb * 32;
    const __bf16* kr = k + ((size_t)bh * T_S + kbase + r) * T_DK;
    v16bf bk0lo = *(const v16bf*)(kr + half * 16);
    v16bf bk0hi = *(const v16bf*)(kr + 32 + half * 16);
    v16bf bk1lo = *(const v16bf*)(kr + 16 * T_DK + half * 16);
    v16bf bk1hi = *(const v16bf*)(kr + 16 * T_DK + 32 + half * 16);
    v8f s0 = {}, s1 = {};
    s0 = wmma_bf16(aq0, bk0lo, s0); s0 = wmma_bf16(aq1, bk0hi, s0);
    s1 = wmma_bf16(aq0, bk1lo, s1); s1 = wmma_bf16(aq1, bk1hi, s1);

#pragma unroll
    for (int i = 0; i < 8; i++) {
      int row = i + 8 * half;
      int qpos = qbase + row;
      float x0 = s0[i] * 0.125f;              // 1/sqrt(64)
      float x1 = s1[i] * 0.125f;
      x0 = (kbase + r      > qpos) ? -3.0e38f : x0;
      x1 = (kbase + 16 + r > qpos) ? -3.0e38f : x1;
      float mx = fmaxf(x0, x1);
      for (int off = 1; off < 16; off <<= 1) mx = fmaxf(mx, __shfl_xor(mx, off, 32));
      float mnew = fmaxf(mrun[i], mx);
      float p0 = __expf(x0 - mnew), p1 = __expf(x1 - mnew);
      float ls = p0 + p1;
      for (int off = 1; off < 16; off <<= 1) ls += __shfl_xor(ls, off, 32);
      float corr = __expf(mrun[i] - mnew);
      lrun[i] = lrun[i] * corr + ls;
      mrun[i] = mnew;
      oacc[0][i] *= corr; oacc[1][i] *= corr; oacc[2][i] *= corr; oacc[3][i] *= corr;
      pl[row * 32 + r]      = (__bf16)p0;     // C-layout -> LDS for A-frag reload
      pl[row * 32 + 16 + r] = (__bf16)p1;
    }
    // intra-wave LDS RAW: DS in-order per wave; drain CDNA5 split DS counter
    asm volatile("s_wait_dscnt 0" ::: "memory");
    v16bf pa = *(const v16bf*)(pl + r * 32 + half * 16);   // A-frag of P (16x32)

    const __bf16* vb = vT + (size_t)bh * T_DK * T_S;
#pragma unroll
    for (int dt = 0; dt < 4; dt++) {          // O += P @ V over DK in 4 N-tiles
      v16bf bvf = *(const v16bf*)(vb + (size_t)(dt * 16 + r) * T_S + kbase + half * 16);
      oacc[dt] = wmma_bf16(pa, bvf, oacc[dt]);
    }
  }

  int b = bh >> 4, hh = bh & 15;
#pragma unroll
  for (int i = 0; i < 8; i++) {
    int row = i + 8 * half;
    float rl = 1.0f / lrun[i];
    int tok = qbase + row;
#pragma unroll
    for (int dt = 0; dt < 4; dt++)
      o[((size_t)b * T_S + tok) * T_D + hh * T_DK + dt * 16 + r] =
          (__bf16)(oacc[dt][i] * rl);
  }
}

// ---------------- SwiGLU: g = u1 * silu(u2), u:[M,2*DFF] bf16 -> g:[M,DFF] ----
__global__ void k_swiglu(const __bf16* __restrict__ u, __bf16* __restrict__ g) {
  size_t idx = (size_t)blockIdx.x * blockDim.x + threadIdx.x;  // M*DFF threads
  size_t m = idx >> 11;                                        // DFF == 2048
  int j = idx & (T_DFF - 1);
  float u1 = (float)u[m * (2 * T_DFF) + j];
  float u2 = (float)u[m * (2 * T_DFF) + T_DFF + j];
  float sig = 1.0f / (1.0f + __expf(-u2));
  g[idx] = (__bf16)(u1 * u2 * sig);
}

// ---------------- host orchestration ------------------------------------------
extern "C" void kernel_launch(void* const* d_in, const int* in_sizes, int n_in,
                              void* d_out, int out_size, void* d_ws, size_t ws_size,
                              hipStream_t stream) {
  (void)in_sizes; (void)n_in; (void)out_size; (void)ws_size;
  const float* x   = (const float*)d_in[0];
  // d_in[1] = causal mask (implicit in kernel)
  const float* wq  = (const float*)d_in[2];
  const float* bq  = (const float*)d_in[3];
  const float* wk  = (const float*)d_in[4];
  const float* bk  = (const float*)d_in[5];
  const float* wv  = (const float*)d_in[6];
  const float* bv  = (const float*)d_in[7];
  const float* wo  = (const float*)d_in[8];
  const float* bo  = (const float*)d_in[9];
  const float* n1w = (const float*)d_in[10];
  const float* n2w = (const float*)d_in[11];
  const float* w1  = (const float*)d_in[12];
  const float* w2  = (const float*)d_in[13];
  float* out = (float*)d_out;

  char* ws = (char*)d_ws;
  size_t off = 0;
  auto alloc = [&](size_t bytes) {
    size_t o = off;
    off = (o + bytes + 255) & ~(size_t)255;
    return o;
  };
  __bf16* wqT  = (__bf16*)(ws + alloc((size_t)T_D * T_D * 2));
  __bf16* wkT  = (__bf16*)(ws + alloc((size_t)T_D * T_D * 2));
  __bf16* wvT  = (__bf16*)(ws + alloc((size_t)T_D * T_D * 2));
  __bf16* woT  = (__bf16*)(ws + alloc((size_t)T_D * T_D * 2));
  __bf16* w1T  = (__bf16*)(ws + alloc((size_t)T_D * 2 * T_DFF * 2));
  __bf16* w2T  = (__bf16*)(ws + alloc((size_t)T_DFF * T_D * 2));
  __bf16* h1   = (__bf16*)(ws + alloc((size_t)T_M * T_D * 2));
  __bf16* qbuf = (__bf16*)(ws + alloc((size_t)T_M * T_D * 2));
  __bf16* kbuf = (__bf16*)(ws + alloc((size_t)T_M * T_D * 2));
  __bf16* vTb  = (__bf16*)(ws + alloc((size_t)T_M * T_D * 2));
  __bf16* obuf = (__bf16*)(ws + alloc((size_t)T_M * T_D * 2));
  float*  x2   = (float*)(ws + alloc((size_t)T_M * T_D * 4));
  __bf16* h2   = (__bf16*)(ws + alloc((size_t)T_M * T_D * 2));
  __bf16* ubuf = (__bf16*)(ws + alloc((size_t)T_M * 2 * T_DFF * 2));
  __bf16* gbuf = (__bf16*)(ws + alloc((size_t)T_M * T_DFF * 2));

  // 1) weights -> bf16, transposed [N,K]
  k_transpose_cvt<<<(T_D * T_D + 255) / 256, 256, 0, stream>>>(wq, wqT, T_D, T_D);
  k_transpose_cvt<<<(T_D * T_D + 255) / 256, 256, 0, stream>>>(wk, wkT, T_D, T_D);
  k_transpose_cvt<<<(T_D * T_D + 255) / 256, 256, 0, stream>>>(wv, wvT, T_D, T_D);
  k_transpose_cvt<<<(T_D * T_D + 255) / 256, 256, 0, stream>>>(wo, woT, T_D, T_D);
  k_transpose_cvt<<<(T_D * 2 * T_DFF + 255) / 256, 256, 0, stream>>>(w1, w1T, T_D, 2 * T_DFF);
  k_transpose_cvt<<<(T_DFF * T_D + 255) / 256, 256, 0, stream>>>(w2, w2T, T_DFF, T_D);

  // 2) norm1
  k_rmsnorm<<<T_M, 256, 0, stream>>>(x, n1w, h1);

  // 3) QKV GEMMs (WMMA), macro-tile 128x64
  dim3 gp(T_D / 64, T_M / 128);
  k_gemm_wmma<<<gp, 256, 0, stream>>>(h1, wqT, bq, nullptr, nullptr, qbuf, T_M, T_D, T_D, 1);
  k_gemm_wmma<<<gp, 256, 0, stream>>>(h1, wkT, bk, nullptr, nullptr, kbuf, T_M, T_D, T_D, 1);
  k_gemm_wmma<<<gp, 256, 0, stream>>>(h1, wvT, bv, nullptr, nullptr, vTb,  T_M, T_D, T_D, 2);

  // 4) RoPE
  k_rope<<<(T_B * T_H * T_S * 32) / 256, 256, 0, stream>>>(qbuf, kbuf);

  // 5) flash attention
  dim3 ga(T_S / 64, T_B * T_H);
  k_attn_flash<<<ga, 128, 0, stream>>>(qbuf, kbuf, vTb, obuf);

  // 6) output projection + residual -> x2 (f32)
  k_gemm_wmma<<<gp, 256, 0, stream>>>(obuf, woT, bo, x, x2, nullptr, T_M, T_D, T_D, 3);

  // 7) norm2
  k_rmsnorm<<<T_M, 256, 0, stream>>>(x2, n2w, h2);

  // 8) FFN up (no bias) -> u
  dim3 gu(2 * T_DFF / 64, T_M / 128);
  k_gemm_wmma<<<gu, 256, 0, stream>>>(h2, w1T, nullptr, nullptr, nullptr, ubuf,
                                      T_M, 2 * T_DFF, T_D, 0);

  // 9) SwiGLU
  k_swiglu<<<((size_t)T_M * T_DFF) / 256, 256, 0, stream>>>(ubuf, gbuf);

  // 10) FFN down + residual -> out (f32)
  k_gemm_wmma<<<gp, 256, 0, stream>>>(gbuf, w2T, nullptr, x2, out, nullptr,
                                      T_M, T_D, T_DFF, 3);
}
